// GemmaAttentionWithExpert_15590731284812
// MI455X (gfx1250) — matrix operand order repaired and checked by reading, compiled, not run on hardware
//
#include <hip/hip_runtime.h>

// ---------------------------------------------------------------------------
// CDNA5 (gfx1250) WMMA types & helpers
// ---------------------------------------------------------------------------
typedef __attribute__((ext_vector_type(16))) __bf16 v16bf;
typedef __attribute__((ext_vector_type(8)))  __bf16 v8bf;
typedef __attribute__((ext_vector_type(8)))  float  v8f;

union Frag16 { v16bf v; v8bf h[2]; };

__device__ __forceinline__ v8f wmma_bf16(v16bf a, v16bf b, v8f c) {
  // D = A(16x32 bf16) * B(32x16 bf16) + C(16x16 f32)
  return __builtin_amdgcn_wmma_f32_16x16x32_bf16(false, a, false, b,
                                                 (short)0, c, false, false);
}

// Async copy 16B per lane: global -> LDS, tracked by ASYNCcnt.
// NOTE: INST_OFFSET is added to BOTH the LDS destination and the global source
// (CDNA5 ISA), so one base covers several chunks.
__device__ __forceinline__ void async_b128(const void* lds_dst, const void* gsrc) {
  unsigned loff = (unsigned)(size_t)lds_dst;
  asm volatile("global_load_async_to_lds_b128 %0, %1, off"
               :: "v"(loff), "v"(gsrc) : "memory");
}
__device__ __forceinline__ void async_b128_off16(const void* lds_dst, const void* gsrc) {
  unsigned loff = (unsigned)(size_t)lds_dst;
  asm volatile("global_load_async_to_lds_b128 %0, %1, off offset:16"
               :: "v"(loff), "v"(gsrc) : "memory");
}
__device__ __forceinline__ void async_b128_off(const void* lds_dst, const void* gsrc, int imm) {
  unsigned loff = (unsigned)(size_t)lds_dst;
  switch (imm) {
    case 16: asm volatile("global_load_async_to_lds_b128 %0, %1, off offset:16" :: "v"(loff), "v"(gsrc) : "memory"); break;
    case 32: asm volatile("global_load_async_to_lds_b128 %0, %1, off offset:32" :: "v"(loff), "v"(gsrc) : "memory"); break;
    case 48: asm volatile("global_load_async_to_lds_b128 %0, %1, off offset:48" :: "v"(loff), "v"(gsrc) : "memory"); break;
    default: asm volatile("global_load_async_to_lds_b128 %0, %1, off"           :: "v"(loff), "v"(gsrc) : "memory"); break;
  }
}
__device__ __forceinline__ void wait_async0() {
  asm volatile("s_wait_asynccnt 0x0" ::: "memory");
}

// ---------------------------------------------------------------------------
// Elementwise convert f32 -> bf16
// ---------------------------------------------------------------------------
__global__ void cvt_f32_bf16(const float* __restrict__ in, __bf16* __restrict__ out, long n) {
  long i = (long)blockIdx.x * blockDim.x + threadIdx.x;
  if (i < n) out[i] = (__bf16)in[i];
}

// W [K][N] f32 -> Wt [N][K] bf16.  N is a power of two: nsh = log2(N).
__global__ void transpose_cvt(const float* __restrict__ in, __bf16* __restrict__ out,
                              int K, int nsh) {
  long i = (long)blockIdx.x * blockDim.x + threadIdx.x;
  long total = (long)K << nsh;
  if (i >= total) return;
  int k = (int)(i >> nsh);
  int n = (int)(i & ((1 << nsh) - 1));
  out[(long)n * K + k] = (__bf16)in[i];
}

// ---------------------------------------------------------------------------
// RoPE (Gemma half-split) on f32 input, bf16 output. HD=256, half=128, L=1024.
// Hn is a power of two: hsh = log2(Hn).
// ---------------------------------------------------------------------------
__global__ void rope_cvt_kernel(const float* __restrict__ in, __bf16* __restrict__ out,
                                const int* __restrict__ pos, int hsh, long total) {
  long i = (long)blockIdx.x * blockDim.x + threadIdx.x;
  if (i >= total) return;
  int j = (int)(i & 127);
  long t = i >> 7;
  int h = (int)(t & ((1 << hsh) - 1)); t >>= hsh;
  int l = (int)(t & 1023);
  int b = (int)(t >> 10);
  float p = (float)pos[b * 1024 + l];
  // 10000^(-j/128) = exp(-j * ln(10000)/128)
  float f = p * __expf(-(float)j * (9.210340371976184f / 128.0f));
  float s = sinf(f), c = cosf(f);
  long base = ((((long)(b * 1024 + l)) << hsh) + h) << 8;
  float x1 = in[base + j], x2 = in[base + 128 + j];
  out[base + j]       = (__bf16)(x1 * c - x2 * s);
  out[base + 128 + j] = (__bf16)(x2 * c + x1 * s);
}

// ---------------------------------------------------------------------------
// Generic bf16 WMMA GEMM:  C[M,N] = A[M,K] * W[K,N],  W given transposed [N][K].
// Block tile 128x128, K-step 32. 8 waves in a 4(M)x2(N) grid, each wave owns a
// 32x64 sub-tile => 8 WMMAs per K-step per wave. A/B tiles staged with
// GLOBAL_LOAD_ASYNC_TO_LDS_B128 (ASYNCcnt), no VGPR round-trip.
//
// Segment remap (pali/expert rows <-> combined [B,1024,.] rows): since TM=128
// divides both segment lengths (768, 256), a block never crosses a segment
// boundary, so the remap collapses to ONE additive constant per block:
//   row' = row + (m0/seg)*(tot-seg) + off          (seg==0 -> identity)
//
// OUT_BF16 is a compile-time parameter so the epilogue has no per-store branch.
// ---------------------------------------------------------------------------
template <bool OUT_BF16>
__global__ __launch_bounds__(256)
void gemm_bf16_wmma(const __bf16* __restrict__ A, const __bf16* __restrict__ Wt,
                    void* __restrict__ Cp,
                    int M, int N, int K,
                    int aseg, int atot, int aoff,
                    int cseg, int ctot, int coff) {
  constexpr int TM = 128, TN = 128, TK = 32;
  __shared__ __attribute__((aligned(32))) __bf16 As[TM * TK];   // 8 KB
  __shared__ __attribute__((aligned(32))) __bf16 Bs[TN * TK];   // 8 KB

  const int tid  = threadIdx.x;
  const int lane = tid & 31;
  const int w    = tid >> 5;     // 0..7
  const int wm   = w >> 1;       // 0..3 (M direction, 32 rows each)
  const int wn   = w & 1;        // 0..1 (N direction, 64 cols each)
  const int l15  = lane & 15;
  const int g    = lane >> 4;
  const int m0   = blockIdx.x * TM;
  const int n0   = blockIdx.y * TN;

  // block-uniform additive row remaps (scalar, one division per kernel)
  const long aAdd = (aseg == 0) ? 0 : (long)(m0 / aseg) * (atot - aseg) + aoff;
  const long cAdd = (cseg == 0) ? 0 : (long)(m0 / cseg) * (ctot - cseg) + coff;

  // staging assignment: each thread owns one 32B chunk of A and one of B
  const int srow  = tid >> 1;           // 0..127
  const int shalf = (tid & 1) * 16;     // element offset 0 / 16
  const __bf16* agp = A  + ((long)(m0 + srow) + aAdd) * (long)K + shalf;
  const __bf16* bgp = Wt + (long)(n0 + srow) * K + shalf;
  __bf16* asp = As + srow * TK + shalf;
  __bf16* bsp = Bs + srow * TK + shalf;

  v8f acc[2][4] = {};

  for (int k0 = 0; k0 < K; k0 += TK) {
    __syncthreads();
    // async global -> LDS staging (32B per thread per matrix)
    async_b128(asp, agp + k0);
    async_b128_off16(asp, agp + k0);
    async_b128(bsp, bgp + k0);
    async_b128_off16(bsp, bgp + k0);
    wait_async0();
    __syncthreads();

#pragma unroll
    for (int mf = 0; mf < 2; ++mf) {
      const int m = wm * 32 + mf * 16 + l15;
      Frag16 a;
      a.h[0] = *(const v8bf*)(As + m * TK + 8 * g);        // K = 8g..8g+7
      a.h[1] = *(const v8bf*)(As + m * TK + 16 + 8 * g);   // K = 16+8g..
#pragma unroll
      for (int nf = 0; nf < 4; ++nf) {
        const int n = wn * 64 + nf * 16 + l15;
        Frag16 b;
        b.h[0] = *(const v8bf*)(Bs + n * TK + 8 * g);        // K = 8g..8g+7
        b.h[1] = *(const v8bf*)(Bs + n * TK + 16 + 8 * g);   // K = 16+8g..
        acc[mf][nf] = wmma_bf16(a.v, b.v, acc[mf][nf]);
      }
    }
  }

  // epilogue: C layout -> lane l elem r: row = r + 8*(l>>4), col = l&15
#pragma unroll
  for (int mf = 0; mf < 2; ++mf)
#pragma unroll
    for (int nf = 0; nf < 4; ++nf) {
      const int col = n0 + wn * 64 + nf * 16 + l15;
#pragma unroll
      for (int r = 0; r < 8; ++r) {
        const int ml = m0 + wm * 32 + mf * 16 + r + 8 * g;
        const long grow = (long)ml + cAdd;
        const float val = acc[mf][nf][r];
        if (OUT_BF16) ((__bf16*)Cp)[grow * (long)N + col] = (__bf16)val;
        else          ((float*)Cp)[grow * (long)N + col] = val;
      }
    }
}

// ---------------------------------------------------------------------------
// Flash attention (online softmax), GQA 1 KV head, B=4 H=8 L=1024 HD=256.
// Grid: (L/128, B*H). 8 waves/block; wave owns 16 query rows x full 256 dims.
// 32-key chunks: QK^T = 16 WMMAs, PV = 16 WMMAs per wave per chunk.
// K tile staged with async-to-LDS; V tile needs a transpose so it stays manual.
// ---------------------------------------------------------------------------
__global__ __launch_bounds__(256)
void flash_attn_wmma(const __bf16* __restrict__ Q, const __bf16* __restrict__ Kb,
                     const __bf16* __restrict__ Vb, const float* __restrict__ mask,
                     __bf16* __restrict__ Out) {
  constexpr int kL = 1024, kHD = 256, kH = 8, kFQ = 128, kFK = 32;
  __shared__ __attribute__((aligned(32))) __bf16 Ks[kFK * kHD];   // 16 KB [key][d]
  __shared__ __attribute__((aligned(32))) __bf16 Vt[kHD * kFK];   // 16 KB [d][key]
  __shared__ __attribute__((aligned(32))) __bf16 Ps[8 * 16 * kFK];//  8 KB per-wave P

  const int tid  = threadIdx.x;
  const int lane = tid & 31;
  const int w    = tid >> 5;
  const int l15  = lane & 15;
  const int g    = lane >> 4;
  const int qb   = blockIdx.x;          // 0..7
  const int bh   = blockIdx.y;          // 0..31
  const int b    = bh >> 3, h = bh & 7;
  const int q0   = qb * kFQ;
  const int qw   = q0 + w * 16;         // wave's first query row

  // Q fragments for this wave's 16 rows, full 256 dims (8 A-frags), in regs.
  Frag16 qf[8];
  {
    const __bf16* qrow = Q + ((long)(b * kL + qw + l15) * kH + h) * kHD;
#pragma unroll
    for (int df = 0; df < 8; ++df) {
      qf[df].h[0] = *(const v8bf*)(qrow + df * 32 + 8 * g);
      qf[df].h[1] = *(const v8bf*)(qrow + df * 32 + 16 + 8 * g);
    }
  }

  v8f O[16] = {};
  float mrow[8], lrow[8];
#pragma unroll
  for (int r = 0; r < 8; ++r) { mrow[r] = -1e30f; lrow[r] = 0.0f; }
  const float scale = 0.0625f;  // 1/sqrt(256)
  const float* mbase = mask + (long)b * kL * kL;

  // per-thread staging assignments
  const int krow = tid >> 3, kc0 = (tid & 7) * 32;     // K tile: 16B x4 per thread
  const int vkey = tid >> 3, vd0 = (tid & 7) * 32;     // V tile: transpose scatter

  for (int kc = 0; kc < kL; kc += kFK) {
    __syncthreads();
    { // stage K chunk rows [32 key][256 d] via async-to-LDS (64B per thread)
      const __bf16* src = Kb + (long)(b * kL + kc + krow) * kHD + kc0;
      __bf16* dst = Ks + krow * kHD + kc0;
      async_b128(dst, src);
      async_b128_off(dst, src, 16);
      async_b128_off(dst, src, 32);
      async_b128_off(dst, src, 48);
      if (kc + kFK < kL) __builtin_prefetch(src + kFK * kHD, 0, 3);
    }
    { // stage V chunk transposed: Vt[d][key]
      const __bf16* src = Vb + (long)(b * kL + kc + vkey) * kHD + vd0;
#pragma unroll
      for (int d = 0; d < 32; ++d) Vt[(vd0 + d) * kFK + vkey] = src[d];
    }
    wait_async0();
    __syncthreads();

    // S = Q * K^T  (2 n-frags x 8 d-frags)
    v8f S[2] = {};
#pragma unroll
    for (int df = 0; df < 8; ++df) {
#pragma unroll
      for (int nf = 0; nf < 2; ++nf) {
        const int n = nf * 16 + l15;   // key within chunk
        Frag16 kfr;
        kfr.h[0] = *(const v8bf*)(Ks + n * kHD + df * 32 + 8 * g);
        kfr.h[1] = *(const v8bf*)(Ks + n * kHD + df * 32 + 16 + 8 * g);
        S[nf] = wmma_bf16(qf[df].v, kfr.v, S[nf]);
      }
    }

    // scale + mask, online softmax (rows live on half-waves: m = r + 8g)
    float sm[2][8], curmax[8];
#pragma unroll
    for (int r = 0; r < 8; ++r) curmax[r] = -1e30f;
#pragma unroll
    for (int nf = 0; nf < 2; ++nf) {
      const int key = kc + nf * 16 + l15;
#pragma unroll
      for (int r = 0; r < 8; ++r) {
        const int qrow = qw + r + 8 * g;
        float s = S[nf][r] * scale + mbase[(long)qrow * kL + key];
        sm[nf][r] = s;
        curmax[r] = fmaxf(curmax[r], s);
      }
    }
#pragma unroll
    for (int r = 0; r < 8; ++r) {
      float v = curmax[r];
      for (int off = 1; off < 16; off <<= 1) v = fmaxf(v, __shfl_xor(v, off, 16));
      curmax[r] = v;
    }
    float alpha[8], psum[8];
#pragma unroll
    for (int r = 0; r < 8; ++r) {
      float mn = fmaxf(mrow[r], curmax[r]);
      alpha[r] = __expf(mrow[r] - mn);
      mrow[r] = mn;
      psum[r] = 0.0f;
    }
#pragma unroll
    for (int nf = 0; nf < 2; ++nf)
#pragma unroll
      for (int r = 0; r < 8; ++r) {
        float p = __expf(sm[nf][r] - mrow[r]);
        psum[r] += p;
        Ps[(w * 16 + r + 8 * g) * kFK + nf * 16 + l15] = (__bf16)p;  // wave-private
      }
#pragma unroll
    for (int r = 0; r < 8; ++r) {
      float v = psum[r];
      for (int off = 1; off < 16; off <<= 1) v += __shfl_xor(v, off, 16);
      lrow[r] = lrow[r] * alpha[r] + v;
    }
#pragma unroll
    for (int d2 = 0; d2 < 16; ++d2)
#pragma unroll
      for (int r = 0; r < 8; ++r) O[d2][r] *= alpha[r];

    // O += P * V  (1 A-frag of 16x32 keys x 16 d-frags)
    Frag16 pf;
    pf.h[0] = *(const v8bf*)(Ps + (w * 16 + l15) * kFK + 8 * g);
    pf.h[1] = *(const v8bf*)(Ps + (w * 16 + l15) * kFK + 16 + 8 * g);
#pragma unroll
    for (int d2 = 0; d2 < 16; ++d2) {
      const int n = d2 * 16 + l15;   // output dim column
      Frag16 vfr;
      vfr.h[0] = *(const v8bf*)(Vt + n * kFK + 8 * g);
      vfr.h[1] = *(const v8bf*)(Vt + n * kFK + 16 + 8 * g);
      O[d2] = wmma_bf16(pf.v, vfr.v, O[d2]);
    }
  }

  // normalize and store bf16 into [B*L][H*HD]
#pragma unroll
  for (int r = 0; r < 8; ++r) lrow[r] = 1.0f / lrow[r];
#pragma unroll
  for (int d2 = 0; d2 < 16; ++d2) {
    const int col = h * kHD + d2 * 16 + l15;
#pragma unroll
    for (int r = 0; r < 8; ++r) {
      const int qrow = qw + r + 8 * g;
      Out[(long)(b * kL + qrow) * 2048 + col] = (__bf16)(O[d2][r] * lrow[r]);
    }
  }
}

// ---------------------------------------------------------------------------
// Host orchestration
// ---------------------------------------------------------------------------
extern "C" void kernel_launch(void* const* d_in, const int* in_sizes, int n_in,
                              void* d_out, int out_size, void* d_ws, size_t ws_size,
                              hipStream_t stream) {
  (void)in_sizes; (void)n_in; (void)out_size; (void)ws_size;
  const float* pali = (const float*)d_in[0];   // [4,768,2048]
  const float* expe = (const float*)d_in[1];   // [4,256,1024]
  const int*   pos  = (const int*)d_in[2];     // [4,1024]
  const float* mask = (const float*)d_in[3];   // [4,1024,1024]
  const float* wq_p = (const float*)d_in[5];
  const float* wk_p = (const float*)d_in[6];
  const float* wv_p = (const float*)d_in[7];
  const float* wo_p = (const float*)d_in[8];
  const float* wq_e = (const float*)d_in[9];
  const float* wk_e = (const float*)d_in[10];
  const float* wv_e = (const float*)d_in[11];
  const float* wo_e = (const float*)d_in[12];

  char* p = (char*)d_ws;
  auto take = [&](size_t bytes) -> char* {
    char* r = p; p += (bytes + 255) & ~(size_t)255; return r;
  };
  __bf16* pe_bf  = (__bf16*)take((size_t)4*768*2048*2);
  __bf16* ee_bf  = (__bf16*)take((size_t)4*256*1024*2);
  __bf16* wqp_t  = (__bf16*)take((size_t)2048*2048*2);
  __bf16* wkp_t  = (__bf16*)take((size_t)256*2048*2);
  __bf16* wvp_t  = (__bf16*)take((size_t)256*2048*2);
  __bf16* wop_t  = (__bf16*)take((size_t)2048*2048*2);
  __bf16* wqe_t  = (__bf16*)take((size_t)2048*1024*2);
  __bf16* wke_t  = (__bf16*)take((size_t)256*1024*2);
  __bf16* wve_t  = (__bf16*)take((size_t)256*1024*2);
  __bf16* woe_t  = (__bf16*)take((size_t)1024*2048*2);
  float*  qf32   = (float*)take((size_t)4*1024*2048*4);
  float*  kf32   = (float*)take((size_t)4*1024*256*4);
  __bf16* q_bf   = (__bf16*)take((size_t)4*1024*2048*2);
  __bf16* k_bf   = (__bf16*)take((size_t)4*1024*256*2);
  __bf16* v_bf   = (__bf16*)take((size_t)4*1024*256*2);
  __bf16* ao_bf  = (__bf16*)take((size_t)4*1024*2048*2);

  const int T = 256;
  auto gr = [&](long n) { return (unsigned)((n + T - 1) / T); };

  // 1) convert embeds to bf16
  cvt_f32_bf16<<<gr(6291456), T, 0, stream>>>(pali, pe_bf, 6291456);
  cvt_f32_bf16<<<gr(1048576), T, 0, stream>>>(expe, ee_bf, 1048576);

  // 2) transpose+convert weights to [N][K] bf16 (nsh = log2(N))
  transpose_cvt<<<gr(4194304), T, 0, stream>>>(wq_p, wqp_t, 2048, 11);
  transpose_cvt<<<gr(524288),  T, 0, stream>>>(wk_p, wkp_t, 2048, 8);
  transpose_cvt<<<gr(524288),  T, 0, stream>>>(wv_p, wvp_t, 2048, 8);
  transpose_cvt<<<gr(4194304), T, 0, stream>>>(wo_p, wop_t, 2048, 11);
  transpose_cvt<<<gr(2097152), T, 0, stream>>>(wq_e, wqe_t, 1024, 11);
  transpose_cvt<<<gr(262144),  T, 0, stream>>>(wk_e, wke_t, 1024, 8);
  transpose_cvt<<<gr(262144),  T, 0, stream>>>(wv_e, wve_t, 1024, 8);
  transpose_cvt<<<gr(2097152), T, 0, stream>>>(wo_e, woe_t, 2048, 10);

  // 3) QKV projections (scatter segment rows into combined [B,1024,.])
  gemm_bf16_wmma<false><<<dim3(24,16), T, 0, stream>>>(pe_bf, wqp_t, qf32,
      3072,2048,2048, 0,0,0, 768,1024,0);
  gemm_bf16_wmma<false><<<dim3(8,16),  T, 0, stream>>>(ee_bf, wqe_t, qf32,
      1024,2048,1024, 0,0,0, 256,1024,768);
  gemm_bf16_wmma<false><<<dim3(24,2),  T, 0, stream>>>(pe_bf, wkp_t, kf32,
      3072,256,2048,  0,0,0, 768,1024,0);
  gemm_bf16_wmma<false><<<dim3(8,2),   T, 0, stream>>>(ee_bf, wke_t, kf32,
      1024,256,1024,  0,0,0, 256,1024,768);
  gemm_bf16_wmma<true><<<dim3(24,2),   T, 0, stream>>>(pe_bf, wvp_t, v_bf,
      3072,256,2048,  0,0,0, 768,1024,0);
  gemm_bf16_wmma<true><<<dim3(8,2),    T, 0, stream>>>(ee_bf, wve_t, v_bf,
      1024,256,1024,  0,0,0, 256,1024,768);

  // 4) RoPE (f32 -> bf16); hsh = log2(heads)
  rope_cvt_kernel<<<gr(4194304), T, 0, stream>>>(qf32, q_bf, pos, 3, 4194304);
  rope_cvt_kernel<<<gr(524288),  T, 0, stream>>>(kf32, k_bf, pos, 0, 524288);

  // 5) fused flash attention
  flash_attn_wmma<<<dim3(8, 32), T, 0, stream>>>(q_bf, k_bf, v_bf, mask, ao_bf);

  // 6) output projections (gather segment rows from combined seq)
  float* out_p = (float*)d_out;
  float* out_e = out_p + (size_t)4*768*2048;
  gemm_bf16_wmma<false><<<dim3(24,16), T, 0, stream>>>(ao_bf, wop_t, out_p,
      3072,2048,2048, 768,1024,0,   0,0,0);
  gemm_bf16_wmma<false><<<dim3(8,8),   T, 0, stream>>>(ao_bf, woe_t, out_e,
      1024,1024,2048, 256,1024,768, 0,0,0);
}
